// AdaAttentionalGNN_89739046683471
// MI455X (gfx1250) — matrix operand order repaired and checked by reading, not compiled, not run on hardware
//
#include <hip/hip_runtime.h>
#include <hip/hip_bf16.h>

// ---------------------------------------------------------------------------
// AdaAttentionalGNN for MI455X (gfx1250): bf16 WMMA GEMMs + f32 softmax/norm,
// exact radix-select top-k pruning. Wave32, v_wmma_f32_16x16x32_bf16 core.
// Round 3: K loop unrolled x2 with ping-pong fragment sets (no register
// rotation moves); loads for one set issue while the other set computes.
// ---------------------------------------------------------------------------

typedef __bf16 bhalf;
typedef __attribute__((ext_vector_type(8)))  __bf16 v8bf;
typedef __attribute__((ext_vector_type(16))) __bf16 v16bf;
typedef __attribute__((ext_vector_type(8)))  float  v8f;

enum GemmMode {
    MODE_F32 = 0,      // out f32 [M,N] row-major, += bias[c]
    MODE_F32_ADD = 1,  // out f32 [M,N] row-major, out += acc + bias[c]  (delta into x)
    MODE_TBF16 = 2,    // out bf16 transposed: out[n*ldc + c]
    MODE_QK = 3,       // out bf16 [H, npts, 64]: out[((c&3)*npts + n)*64 + (c>>2)]
    MODE_V = 4,        // out bf16 [H, 64, npts]: out[((c&3)*64 + (c>>2))*npts + n]
    MODE_ATT = 5,      // out bf16 [n, 256]: out[n*256 + c*4 + h]
    MODE_SCORES = 6    // out f32 [M,N]: acc * 0.125 * mask[c*npts + n]
};

// Generic 16x16-tile WMMA GEMM:  C = A(MxK, row-major) * B(KxN)  with B given
// transposed/contiguous as Bt (N x K, row-major).  TPW adjacent N-tiles per
// wave share one A fragment; 4 waves per workgroup.  K multiple of 64.
template <int TPW>
__global__ __launch_bounds__(128) void gemm_wmma(
    const bhalf* __restrict__ A, int lda,
    const bhalf* __restrict__ Bt, int ldb,
    const float* __restrict__ bias,
    const bhalf* __restrict__ mask,
    void* __restrict__ out, int ldc,
    int mode, int hparam, int npts, int Kdim)
{
    const int lane = threadIdx.x & 31;
    const int wave = threadIdx.x >> 5;
    const int tN0  = (blockIdx.x * 4 + wave) * TPW;
    const int tM   = blockIdx.y;
    const int half = lane >> 4;
    const int l15  = lane & 15;
    const int mrow = tM * 16 + l15;     // A-fragment row for this lane

    const bhalf* ap = A + (size_t)mrow * lda + half * 8;   // K-contig, 16B aligned
    const bhalf* bp[TPW];
    int ncol[TPW];
#pragma unroll
    for (int t = 0; t < TPW; ++t) {
        ncol[t] = (tN0 + t) * 16 + l15;                    // B-fragment column
        bp[t] = Bt + (size_t)ncol[t] * ldb + half * 8;     // K-contig, 16B aligned
    }

    v8f acc[TPW];
#pragma unroll
    for (int t = 0; t < TPW; ++t) acc[t] = (v8f){0.f,0.f,0.f,0.f,0.f,0.f,0.f,0.f};

    // ---- ping-pong pipeline: set X = tile k, set Y = tile k+32 ------------
    v8bf xa0 = *(const v8bf*)(ap);
    v8bf xa1 = *(const v8bf*)(ap + 16);
    v8bf xb0[TPW], xb1[TPW];
#pragma unroll
    for (int t = 0; t < TPW; ++t) {
        xb0[t] = *(const v8bf*)(bp[t]);
        xb1[t] = *(const v8bf*)(bp[t] + 16);
    }

    for (int k0 = 0; k0 < Kdim; k0 += 64) {
        const int ky = k0 + 32;
        // issue set-Y loads (tile k+32) before consuming set X
        v8bf ya0 = *(const v8bf*)(ap + ky);
        v8bf ya1 = *(const v8bf*)(ap + ky + 16);
        v8bf yb0[TPW], yb1[TPW];
#pragma unroll
        for (int t = 0; t < TPW; ++t) {
            yb0[t] = *(const v8bf*)(bp[t] + ky);
            yb1[t] = *(const v8bf*)(bp[t] + ky + 16);
        }
        if (k0 + 128 <= Kdim) {         // prefetch two tiles ahead
            __builtin_prefetch(ap + k0 + 128, 0, 1);
            __builtin_prefetch(bp[0] + k0 + 128, 0, 1);
        }
        // WMMAs on set X
        {
            v16bf af;
#pragma unroll
            for (int i = 0; i < 8; ++i) { af[i] = xa0[i]; af[i + 8] = xa1[i]; }
#pragma unroll
            for (int t = 0; t < TPW; ++t) {
                v16bf bfv;
#pragma unroll
                for (int i = 0; i < 8; ++i) { bfv[i] = xb0[t][i]; bfv[i + 8] = xb1[t][i]; }
                acc[t] = __builtin_amdgcn_wmma_f32_16x16x32_bf16(
                             false, af, false, bfv, (short)0, acc[t], false, false);
            }
        }
        // reload set X for tile k+64 while set Y computes (uniform guard)
        if (k0 + 64 < Kdim) {
            const int kx = k0 + 64;
            xa0 = *(const v8bf*)(ap + kx);
            xa1 = *(const v8bf*)(ap + kx + 16);
#pragma unroll
            for (int t = 0; t < TPW; ++t) {
                xb0[t] = *(const v8bf*)(bp[t] + kx);
                xb1[t] = *(const v8bf*)(bp[t] + kx + 16);
            }
        }
        // WMMAs on set Y
        {
            v16bf af;
#pragma unroll
            for (int i = 0; i < 8; ++i) { af[i] = ya0[i]; af[i + 8] = ya1[i]; }
#pragma unroll
            for (int t = 0; t < TPW; ++t) {
                v16bf bfv;
#pragma unroll
                for (int i = 0; i < 8; ++i) { bfv[i] = yb0[t][i]; bfv[i + 8] = yb1[t][i]; }
                acc[t] = __builtin_amdgcn_wmma_f32_16x16x32_bf16(
                             false, af, false, bfv, (short)0, acc[t], false, false);
            }
        }
    }

    const int rbase = tM * 16 + half * 8;
#pragma unroll
    for (int t = 0; t < TPW; ++t) {
        const int nc = ncol[t];
#pragma unroll
        for (int r = 0; r < 8; ++r) {
            const int c = rbase + r;
            float v = acc[t][r] + (bias ? bias[c] : 0.0f);
            switch (mode) {
            case MODE_F32:
                ((float*)out)[(size_t)c * ldc + nc] = v; break;
            case MODE_F32_ADD:
                ((float*)out)[(size_t)c * ldc + nc] += v; break;
            case MODE_TBF16:
                ((bhalf*)out)[(size_t)nc * ldc + c] = (bhalf)v; break;
            case MODE_QK:
                ((bhalf*)out)[((size_t)(c & 3) * npts + nc) * 64 + (c >> 2)] = (bhalf)v; break;
            case MODE_V:
                ((bhalf*)out)[((size_t)((c & 3) * 64 + (c >> 2))) * npts + nc] = (bhalf)v; break;
            case MODE_ATT:
                ((bhalf*)out)[(size_t)nc * 256 + c * 4 + hparam] = (bhalf)v; break;
            case MODE_SCORES: {
                float mv = (float)mask[(size_t)c * npts + nc];
                ((float*)out)[(size_t)c * ldc + nc] = v * 0.125f * mv;
            } break;
            }
        }
    }
}

// Row softmax over 2048 f32 logits -> bf16 probs + head-averaged f32 probs.
__global__ __launch_bounds__(256) void softmax_rows(
    const float* __restrict__ s, bhalf* __restrict__ p,
    float* __restrict__ pa, int accum)
{
    __shared__ float red[256];
    const int row = blockIdx.x, t = threadIdx.x;
    const float* sr = s + (size_t)row * 2048;
    float v[8];
    float mx = -3.402823466e38f;
#pragma unroll
    for (int j = 0; j < 8; ++j) { v[j] = sr[t + j * 256]; mx = fmaxf(mx, v[j]); }
    red[t] = mx; __syncthreads();
    for (int off = 128; off > 0; off >>= 1) {
        if (t < off) red[t] = fmaxf(red[t], red[t + off]);
        __syncthreads();
    }
    mx = red[0]; __syncthreads();
    float se = 0.f;
#pragma unroll
    for (int j = 0; j < 8; ++j) { v[j] = expf(v[j] - mx); se += v[j]; }
    red[t] = se; __syncthreads();
    for (int off = 128; off > 0; off >>= 1) {
        if (t < off) red[t] += red[t + off];
        __syncthreads();
    }
    const float inv = 1.0f / red[0];
#pragma unroll
    for (int j = 0; j < 8; ++j) {
        size_t idx = (size_t)row * 2048 + t + j * 256;
        float pr = v[j] * inv;
        p[idx] = (bhalf)pr;
        float q = 0.25f * pr;           // mean over H=4 heads
        pa[idx] = accum ? (pa[idx] + q) : q;
    }
}

// InstanceNorm1d(affine=False) over N=2048 + ReLU, write bf16 transposed.
__global__ __launch_bounds__(256) void instnorm_relu_T(
    const float* __restrict__ pre, bhalf* __restrict__ hT)
{
    __shared__ float r1[256], r2[256];
    const int c = blockIdx.x, t = threadIdx.x;
    const float* row = pre + (size_t)c * 2048;
    float v[8], s = 0.f, s2 = 0.f;
#pragma unroll
    for (int j = 0; j < 8; ++j) { v[j] = row[t + j * 256]; s += v[j]; s2 += v[j] * v[j]; }
    r1[t] = s; r2[t] = s2; __syncthreads();
    for (int off = 128; off > 0; off >>= 1) {
        if (t < off) { r1[t] += r1[t + off]; r2[t] += r2[t + off]; }
        __syncthreads();
    }
    const float mu = r1[0] * (1.f / 2048.f);
    const float var = r2[0] * (1.f / 2048.f) - mu * mu;
    const float rs = rsqrtf(var + 1e-5f);
#pragma unroll
    for (int j = 0; j < 8; ++j) {
        float h = fmaxf((v[j] - mu) * rs, 0.f);
        hT[(size_t)(t + j * 256) * 512 + c] = (bhalf)h;
    }
}

// Exact per-row top-k via 4-pass radix select on float bits (values >= 0),
// JAX-stable tie break (lower index first); applies (val > 0) filter.
__global__ __launch_bounds__(256) void prune_rows(
    const float* __restrict__ pa, bhalf* __restrict__ mask, int kkeep)
{
    __shared__ unsigned hist[256];
    __shared__ unsigned sh_prefix, sh_rem;
    const int row = blockIdx.x, b = blockIdx.y, t = threadIdx.x;
    const float* pr = pa + ((size_t)b * 2048 + row) * 2048;
    bhalf* mr = mask + ((size_t)b * 2048 + row) * 2048;

    unsigned u[8];
#pragma unroll
    for (int j = 0; j < 8; ++j) u[j] = __float_as_uint(pr[t * 8 + j]);

    if (t == 0) { sh_prefix = 0u; sh_rem = (unsigned)kkeep; }
    __syncthreads();

    for (int pass = 0; pass < 4; ++pass) {
        const int shift = 24 - pass * 8;
        hist[t] = 0u;
        __syncthreads();
        const unsigned pref = sh_prefix;
#pragma unroll
        for (int j = 0; j < 8; ++j) {
            bool match = (pass == 0) || ((u[j] >> (shift + 8)) == (pref >> (shift + 8)));
            if (match) atomicAdd(&hist[(u[j] >> shift) & 255u], 1u);
        }
        __syncthreads();
        if (t == 0) {
            unsigned rem = sh_rem, chosen = 0u;
            for (int bin = 255; bin >= 0; --bin) {
                unsigned c = hist[bin];
                if (c >= rem) { chosen = (unsigned)bin; break; }
                rem -= c;
            }
            sh_prefix = pref | (chosen << shift);
            sh_rem = rem;
        }
        __syncthreads();
    }
    const unsigned T = sh_prefix;
    const unsigned need_eq = sh_rem;
    __syncthreads();

    unsigned lcnt = 0;
#pragma unroll
    for (int j = 0; j < 8; ++j) if (u[j] == T) lcnt++;
    hist[t] = lcnt; __syncthreads();
    for (int off = 1; off < 256; off <<= 1) {
        unsigned x = (t >= off) ? hist[t - off] : 0u;
        __syncthreads();
        hist[t] += x;
        __syncthreads();
    }
    unsigned rank = hist[t] - lcnt;   // equals in lower-index positions
#pragma unroll
    for (int j = 0; j < 8; ++j) {
        const bool eq = (u[j] == T);
        const bool sel = (u[j] > T) || (eq && rank < need_eq);
        if (eq) rank++;
        mr[t * 8 + j] = (bhalf)((sel && u[j] != 0u) ? 1.0f : 0.0f);
    }
}

__global__ void cvt_weights(const float* __restrict__ in, bhalf* __restrict__ out, int n)
{
    int i = blockIdx.x * 256 + threadIdx.x;
    if (i < n) out[i] = (bhalf)in[i];
}

// x f32 [B,256,2048] -> yT bf16 [B,2048,512] columns 0..255 (transpose + cvt)
__global__ void cvt_xT(const float* __restrict__ x, bhalf* __restrict__ yT)
{
    int i = blockIdx.x * 256 + threadIdx.x;      // 0 .. B*256*2048-1
    int n = i & 2047, c = (i >> 11) & 255, b = i >> 19;
    yT[((size_t)b * 2048 + n) * 512 + c] = (bhalf)x[i];
}

__global__ void fill_ones_bf16(bhalf* __restrict__ p, size_t n)
{
    size_t i = (size_t)blockIdx.x * 256 + threadIdx.x;
    if (i < n) p[i] = (bhalf)1.0f;
}

// ---------------------------------------------------------------------------

extern "C" void kernel_launch(void* const* d_in, const int* in_sizes, int n_in,
                              void* d_out, int out_size, void* d_ws, size_t ws_size,
                              hipStream_t stream)
{
    (void)in_sizes; (void)n_in; (void)out_size; (void)ws_size;
    constexpr int Bn = 2, Dd = 256, Nn = 2048, Hh = 4, Ll = 6;
    constexpr size_t NN = (size_t)Nn * Nn;

    const float* desc0 = (const float*)d_in[0];
    const float* desc1 = (const float*)d_in[1];
    const float* Wq = (const float*)d_in[2];  const float* bq = (const float*)d_in[3];
    const float* Wk = (const float*)d_in[4];  const float* bk = (const float*)d_in[5];
    const float* Wv = (const float*)d_in[6];  const float* bv = (const float*)d_in[7];
    const float* Wm = (const float*)d_in[8];  const float* bm = (const float*)d_in[9];
    const float* W1 = (const float*)d_in[10]; const float* b1 = (const float*)d_in[11];
    const float* W2 = (const float*)d_in[12]; const float* b2 = (const float*)d_in[13];

    char* ws = (char*)d_ws;
    size_t off = 0;
    auto alloc = [&](size_t bytes) -> char* {
        char* p = ws + off;
        off = (off + bytes + 255) & ~(size_t)255;
        return p;
    };

    bhalf* wq_b = (bhalf*)alloc((size_t)Ll * Dd * Dd * 2);
    bhalf* wk_b = (bhalf*)alloc((size_t)Ll * Dd * Dd * 2);
    bhalf* wv_b = (bhalf*)alloc((size_t)Ll * Dd * Dd * 2);
    bhalf* wm_b = (bhalf*)alloc((size_t)Ll * Dd * Dd * 2);
    bhalf* w1_b = (bhalf*)alloc((size_t)Ll * 512 * 512 * 2);
    bhalf* w2_b = (bhalf*)alloc((size_t)Ll * 256 * 512 * 2);
    float* x0   = (float*)alloc((size_t)Bn * Dd * Nn * 4);
    float* x1   = (float*)alloc((size_t)Bn * Dd * Nn * 4);
    bhalf* yT0  = (bhalf*)alloc((size_t)Bn * Nn * 512 * 2);
    bhalf* yT1  = (bhalf*)alloc((size_t)Bn * Nn * 512 * 2);
    bhalf* q_hs = (bhalf*)alloc((size_t)Hh * Nn * 64 * 2);   // per-batch reuse
    bhalf* k_hs = (bhalf*)alloc((size_t)Hh * Nn * 64 * 2);
    bhalf* v_hs = (bhalf*)alloc((size_t)Hh * 64 * Nn * 2);
    bhalf* attT = (bhalf*)alloc((size_t)Nn * Dd * 2);
    float* pre  = (float*)alloc((size_t)512 * Nn * 4);
    bhalf* hT   = (bhalf*)alloc((size_t)Nn * 512 * 2);
    float* scores  = (float*)alloc(NN * 4);
    bhalf* probbuf = (bhalf*)alloc(NN * 2);
    float* pas     = (float*)alloc(4 * (size_t)Bn * NN * 4);
    bhalf* masks   = (bhalf*)alloc(4 * (size_t)Bn * NN * 2);

    float* pa00 = pas + 0 * (size_t)Bn * NN;
    float* pa01 = pas + 1 * (size_t)Bn * NN;
    float* pa11 = pas + 2 * (size_t)Bn * NN;
    float* pa10 = pas + 3 * (size_t)Bn * NN;
    bhalf* m00 = masks + 0 * (size_t)Bn * NN;
    bhalf* m01 = masks + 1 * (size_t)Bn * NN;
    bhalf* m11 = masks + 2 * (size_t)Bn * NN;
    bhalf* m10 = masks + 3 * (size_t)Bn * NN;

    // --- one-time (per call) prep: weights -> bf16, x copies, masks = 1 ----
    cvt_weights<<<(Ll * Dd * Dd + 255) / 256, 256, 0, stream>>>(Wq, wq_b, Ll * Dd * Dd);
    cvt_weights<<<(Ll * Dd * Dd + 255) / 256, 256, 0, stream>>>(Wk, wk_b, Ll * Dd * Dd);
    cvt_weights<<<(Ll * Dd * Dd + 255) / 256, 256, 0, stream>>>(Wv, wv_b, Ll * Dd * Dd);
    cvt_weights<<<(Ll * Dd * Dd + 255) / 256, 256, 0, stream>>>(Wm, wm_b, Ll * Dd * Dd);
    cvt_weights<<<(Ll * 512 * 512 + 255) / 256, 256, 0, stream>>>(W1, w1_b, Ll * 512 * 512);
    cvt_weights<<<(Ll * 256 * 512 + 255) / 256, 256, 0, stream>>>(W2, w2_b, Ll * 256 * 512);
    hipMemcpyAsync(x0, desc0, (size_t)Bn * Dd * Nn * 4, hipMemcpyDeviceToDevice, stream);
    hipMemcpyAsync(x1, desc1, (size_t)Bn * Dd * Nn * 4, hipMemcpyDeviceToDevice, stream);
    {
        size_t nm = 4 * (size_t)Bn * NN;
        fill_ones_bf16<<<(unsigned)((nm + 255) / 256), 256, 0, stream>>>(masks, nm);
    }

    // TPW=2 (shared A fragment) for large-M GEMMs; TPW=1 keeps the small
    // PV GEMM (M=64) at 512 waves to fill the WGPs.
    auto gemm = [&](const bhalf* A, int lda, const bhalf* Bt, int ldb,
                    const float* bias, const bhalf* mk, void* out, int ldc,
                    int mode, int hparam, int npts, int Mdim, int Ndim, int Kdim) {
        if (Mdim >= 256) {
            dim3 g(Ndim / 128, Mdim / 16);
            gemm_wmma<2><<<g, dim3(128), 0, stream>>>(A, lda, Bt, ldb, bias, mk,
                                                      out, ldc, mode, hparam, npts, Kdim);
        } else {
            dim3 g(Ndim / 64, Mdim / 16);
            gemm_wmma<1><<<g, dim3(128), 0, stream>>>(A, lda, Bt, ldb, bias, mk,
                                                      out, ldc, mode, hparam, npts, Kdim);
        }
    };

    // One AdaAttentionalPropagation call: x += MLP(concat(x, merge(attn(x,src))))
    auto attn = [&](int li, float* x, bhalf* yT_x, bhalf* yT_s, bhalf* maskk, float* pak) {
        const bhalf* wq_l = wq_b + (size_t)li * Dd * Dd;
        const bhalf* wk_l = wk_b + (size_t)li * Dd * Dd;
        const bhalf* wv_l = wv_b + (size_t)li * Dd * Dd;
        const bhalf* wm_l = wm_b + (size_t)li * Dd * Dd;
        const bhalf* w1_l = w1_b + (size_t)li * 512 * 512;
        const bhalf* w2_l = w2_b + (size_t)li * 256 * 512;
        const float* bq_l = bq + li * Dd;  const float* bk_l = bk + li * Dd;
        const float* bv_l = bv + li * Dd;  const float* bm_l = bm + li * Dd;
        const float* b1_l = b1 + li * 512; const float* b2_l = b2 + li * 256;

        for (int b = 0; b < Bn; ++b) {
            bhalf* yx = yT_x + (size_t)b * Nn * 512;
            bhalf* ys = yT_s + (size_t)b * Nn * 512;
            // Q/K/V projections (K=256) -> head-split bf16 layouts
            gemm(wq_l, Dd, yx, 512, bq_l, nullptr, q_hs, 0, MODE_QK, 0, Nn, Dd, Nn, Dd);
            gemm(wk_l, Dd, ys, 512, bk_l, nullptr, k_hs, 0, MODE_QK, 0, Nn, Dd, Nn, Dd);
            gemm(wv_l, Dd, ys, 512, bv_l, nullptr, v_hs, 0, MODE_V,  0, Nn, Dd, Nn, Dd);
            for (int h = 0; h < Hh; ++h) {
                // scores = (q^T k)/8 * mask   [2048 x 2048], K = 64
                gemm(q_hs + (size_t)h * Nn * 64, 64,
                     k_hs + (size_t)h * Nn * 64, 64,
                     nullptr, maskk + (size_t)b * NN,
                     scores, Nn, MODE_SCORES, 0, Nn, Nn, Nn, 64);
                softmax_rows<<<Nn, 256, 0, stream>>>(scores, probbuf,
                                                     pak + (size_t)b * NN, h > 0);
                // att_h[64, n] = V[64, m] * P^T[m, n], K = 2048
                gemm(v_hs + (size_t)h * 64 * Nn, Nn,
                     probbuf, Nn, nullptr, nullptr,
                     attT, 0, MODE_ATT, h, 0, 64, Nn, Nn);
            }
            // message = Wm * att + bm  -> yT[:, 256:512] (bf16 transposed)
            gemm(wm_l, Dd, attT, Dd, bm_l, nullptr, yx + 256, 512,
                 MODE_TBF16, 0, 0, Dd, Nn, Dd);
            // hidden = W1 * concat(x, message) + b1  (f32)
            gemm(w1_l, 512, yx, 512, b1_l, nullptr, pre, Nn,
                 MODE_F32, 0, 0, 512, Nn, 512);
            instnorm_relu_T<<<512, 256, 0, stream>>>(pre, hT);
            // x += W2 * relu(instnorm(hidden)) + b2
            gemm(w2_l, 512, hT, 512, b2_l, nullptr, x + (size_t)b * Dd * Nn, Nn,
                 MODE_F32_ADD, 0, 0, Dd, Nn, 512);
        }
    };

    auto prune = [&](bhalf* maskk, const float* pak, int knew) {
        prune_rows<<<dim3(Nn, Bn), 256, 0, stream>>>(pak, maskk, knew);
    };

    const int pools[Ll]    = {1, 1, 2, 2, 2, 2};
    const int is_cross[Ll] = {0, 1, 0, 1, 0, 1};
    int k00 = Nn, k11 = Nn, k01 = Nn, k10 = Nn;

    for (int i = 0; i < Ll; ++i) {
        // snapshot current descriptors (bf16, transposed) before any update
        cvt_xT<<<(Bn * Dd * Nn) / 256, 256, 0, stream>>>(x0, yT0);
        cvt_xT<<<(Bn * Dd * Nn) / 256, 256, 0, stream>>>(x1, yT1);
        if (is_cross[i]) {
            if (pools[i] > 1) { k01 /= pools[i]; prune(m01, pa01, k01); }
            attn(i, x0, yT0, yT1, m01, pa01);
            if (pools[i] > 1) { k10 /= pools[i]; prune(m10, pa10, k10); }
            attn(i, x1, yT1, yT0, m10, pa10);
        } else {
            if (pools[i] > 1) { k00 /= pools[i]; prune(m00, pa00, k00); }
            attn(i, x0, yT0, yT0, m00, pa00);
            if (pools[i] > 1) { k11 /= pools[i]; prune(m11, pa11, k11); }
            attn(i, x1, yT1, yT1, m11, pa11);
        }
    }

    // output = stack([desc0, desc1])  -> [2, B, D, N] f32
    hipMemcpyAsync(d_out, x0, (size_t)Bn * Dd * Nn * 4, hipMemcpyDeviceToDevice, stream);
    hipMemcpyAsync((float*)d_out + (size_t)Bn * Dd * Nn, x1,
                   (size_t)Bn * Dd * Nn * 4, hipMemcpyDeviceToDevice, stream);
}
